// MultiHeadAttention_27298812134177
// MI455X (gfx1250) — matrix operand order repaired and checked by reading, not compiled, not run on hardware
//
#include <hip/hip_runtime.h>
#include <hip/hip_bf16.h>

// ---- problem constants -----------------------------------------------------
#define B 4
#define S 2048
#define D 768
#define H 12
#define HD 64

typedef __bf16 bf16;
typedef __attribute__((ext_vector_type(16))) __bf16 v16bf;
typedef __attribute__((ext_vector_type(8)))  __bf16 v8bf;
typedef __attribute__((ext_vector_type(8)))  float   v8f;
typedef __attribute__((ext_vector_type(4)))  unsigned int u32x4;
typedef __attribute__((ext_vector_type(8)))  unsigned int u32x8;

static __device__ __forceinline__ bf16 f2bf(float f) {
    union { float f; unsigned u; } v; v.f = f;
    unsigned r = v.u + 0x7FFFu + ((v.u >> 16) & 1u);   // round-to-nearest-even
    unsigned short h = (unsigned short)(r >> 16);
    return __builtin_bit_cast(bf16, h);
}

static __device__ __forceinline__ v8f vzero8() {
    v8f z = {0.f,0.f,0.f,0.f,0.f,0.f,0.f,0.f};
    return z;
}

static __device__ __forceinline__ v8f wmma_bf16(v16bf a, v16bf b, v8f c) {
    return __builtin_amdgcn_wmma_f32_16x16x32_bf16(
        false, a, false, b, (short)0, c, false, false);
}

// A-fragment (16x32 bf16, row-major source). `p` already includes the lane's
// row offset and the (lane>>4)*16 K-offset (contiguous k = 16h + s).
static __device__ __forceinline__ v16bf load_a_frag(const bf16* p) {
    v8bf lo = *(const v8bf*)p;
    v8bf hi = *(const v8bf*)(p + 8);
    v16bf a;
#pragma unroll
    for (int i = 0; i < 8; ++i) { a[i] = lo[i]; a[i + 8] = hi[i]; }
    return a;
}

// ---- Tensor Data Mover: 2D tile load, global -> LDS (ISA ch.8 D#) ---------
// Elements are 2 bytes (data_size=1). Dims/strides in elements.
static __device__ __forceinline__ void tdm_load_2d(unsigned lds_off,
                                                   const void* gptr,
                                                   unsigned tensor_d0,
                                                   unsigned tensor_d1,
                                                   unsigned tile_d0,
                                                   unsigned tile_d1,
                                                   unsigned long long stride0) {
    unsigned long long ga = (unsigned long long)gptr;
    u32x4 g0;
    g0[0] = 1u;                                             // count=1, no gather
    g0[1] = lds_off;                                        // lds_addr (bytes)
    g0[2] = (unsigned)(ga & 0xFFFFFFFFu);                   // global_addr[31:0]
    g0[3] = (unsigned)((ga >> 32) & 0x01FFFFFFu) | 0x80000000u; // [56:32] | type=2
    u32x8 g1;
    g1[0] = 0x00010000u;                                    // data_size=1 (2B)
    g1[1] = (tensor_d0 & 0xFFFFu) << 16;                    // tensor_dim0[15:0]
    g1[2] = (tensor_d0 >> 16) | ((tensor_d1 & 0xFFFFu) << 16);
    g1[3] = (tensor_d1 >> 16) | ((tile_d0 & 0xFFFFu) << 16);
    g1[4] = tile_d1 & 0xFFFFu;                              // tile_dim2 = 0
    g1[5] = (unsigned)(stride0 & 0xFFFFFFFFu);              // dim0_stride lo
    g1[6] = (unsigned)((stride0 >> 32) & 0xFFFFu);          // dim0_stride hi
    g1[7] = 0u;                                             // dim1_stride unused
    asm volatile("tensor_load_to_lds %0, %1" :: "s"(g0), "s"(g1) : "memory");
}

static __device__ __forceinline__ unsigned lds_off_of(const void* p) {
    return (unsigned)(unsigned long long)p;  // generic LDS addr low 32 = offset
}

// ---- kernel 1: convert inputs to bf16, GEMM-friendly layouts ---------------
__global__ void mha_convert_kernel(const float* __restrict__ X,
                                   const float* __restrict__ Wq,
                                   const float* __restrict__ Wk,
                                   const float* __restrict__ Wv,
                                   const float* __restrict__ Wo,
                                   bf16* __restrict__ Xbf,
                                   bf16* __restrict__ Wqt,
                                   bf16* __restrict__ Wkt,
                                   bf16* __restrict__ Wvt,
                                   bf16* __restrict__ Wob) {
    const size_t nX  = (size_t)B * S * D;
    const size_t nW  = (size_t)H * D * HD;
    const size_t nWo = (size_t)D * D;
    size_t i0     = (size_t)blockIdx.x * blockDim.x + threadIdx.x;
    size_t stride = (size_t)gridDim.x * blockDim.x;

    for (size_t t = i0; t < nX; t += stride) Xbf[t] = f2bf(X[t]);
    for (size_t t = i0; t < nW; t += stride) {           // [H,D,HD]->[H,HD,D]
        int h   = (int)(t / (D * HD));
        int rem = (int)(t % (D * HD));
        int d   = rem / HD;
        int e   = rem % HD;
        size_t o = (size_t)h * HD * D + (size_t)e * D + d;
        Wqt[o] = f2bf(Wq[t]);
        Wkt[o] = f2bf(Wk[t]);
        Wvt[o] = f2bf(Wv[t]);
    }
    for (size_t t = i0; t < nWo; t += stride) Wob[t] = f2bf(Wo[t]);
}

// ---- kernel 2: fused QKV projection (TDM-staged weights in LDS) ------------
// Block = 8 waves, all same (b,h); wave `wid` does a 16-row tile. TDM double-
// buffers the 64x32 weight k-slices for Wq/Wk/Wv.
__global__ void mha_qkv_kernel(const bf16* __restrict__ Xbf,
                               const bf16* __restrict__ Wqt,
                               const bf16* __restrict__ Wkt,
                               const bf16* __restrict__ Wvt,
                               const float* __restrict__ bq,
                               const float* __restrict__ bk,
                               const float* __restrict__ bv,
                               bf16* __restrict__ Qb,
                               bf16* __restrict__ Kb,
                               bf16* __restrict__ Vt) {
    __shared__ __align__(32) bf16 Ws[3][2][64 * 32];     // 24 KB

    const int lane = threadIdx.x & 31;
    const int wid  = threadIdx.x >> 5;
    const int w    = blockIdx.x * 8 + wid;     // 0..6143 ; 128 tiles per (b,h)
    const int bh   = w >> 7;
    const int mt   = w & 127;
    const int b    = bh / H;
    const int h    = bh % H;
    const int s0   = mt * 16;

    const bf16* xrow = Xbf + ((size_t)(b * S + s0 + (lane & 15))) * D + ((lane >> 4) * 16);
    const bf16* wbase[3] = { Wqt + (size_t)h * HD * D,
                             Wkt + (size_t)h * HD * D,
                             Wvt + (size_t)h * HD * D };

    v8f acc[3][4];
#pragma unroll
    for (int t = 0; t < 3; ++t)
#pragma unroll
        for (int j = 0; j < 4; ++j) acc[t][j] = vzero8();

    if (wid == 0) {
#pragma unroll
        for (int t = 0; t < 3; ++t)
            tdm_load_2d(lds_off_of(&Ws[t][0][0]), wbase[t],
                        D, HD, 32, 64, (unsigned long long)D);
    }

#pragma unroll 1
    for (int kk = 0; kk < D; kk += 32) {
        const int buf = (kk >> 5) & 1;
        if (wid == 0) __builtin_amdgcn_s_wait_tensorcnt(0);
        __syncthreads();
        if (wid == 0 && (kk + 32) < D) {
#pragma unroll
            for (int t = 0; t < 3; ++t)
                tdm_load_2d(lds_off_of(&Ws[t][buf ^ 1][0]), wbase[t] + kk + 32,
                            D, HD, 32, 64, (unsigned long long)D);
        }

        v16bf a = load_a_frag(xrow + kk);
        const int fo = (lane & 15) * 32 + ((lane >> 4) * 16);
#pragma unroll
        for (int t = 0; t < 3; ++t) {
            const bf16* wl = &Ws[t][buf][0];
            v16bf bf[4];
#pragma unroll
            for (int j = 0; j < 4; ++j) bf[j] = *(const v16bf*)(wl + j * 16 * 32 + fo);
#pragma unroll
            for (int j = 0; j < 4; ++j) acc[t][j] = wmma_bf16(a, bf[j], acc[t][j]);
        }
    }

    const float qscale = 0.125f;   // 1/sqrt(64)
#pragma unroll
    for (int j = 0; j < 4; ++j) {
        const int e   = j * 16 + (lane & 15);
        const float bqv = bq[h * HD + e];
        const float bkv = bk[h * HD + e];
        const float bvv = bv[h * HD + e];
#pragma unroll
        for (int r = 0; r < 8; ++r) {
            const int m = r + 8 * (lane >> 4);
            const size_t qo = ((size_t)bh * S + s0 + m) * HD + e;
            Qb[qo] = f2bf((acc[0][j][r] + bqv) * qscale);
            Kb[qo] = f2bf(acc[1][j][r] + bkv);
            Vt[(size_t)bh * HD * S + (size_t)e * S + s0 + m] = f2bf(acc[2][j][r] + bvv);
        }
    }
}

// ---- kernel 3: flash attention, TDM-staged K/V tiles -----------------------
// Block = 8 waves, all same (b,h); wave handles a 16-query tile. K tile
// (32x64) and V^T tile (64x32) double-buffered in LDS via TDM.
__global__ void mha_attn_kernel(const bf16* __restrict__ Qb,
                                const bf16* __restrict__ Kb,
                                const bf16* __restrict__ Vt,
                                bf16* __restrict__ AttO) {
    __shared__ __align__(32) bf16 Kt[2][32 * 64];        // 8 KB
    __shared__ __align__(32) bf16 Vs[2][64 * 32];        // 8 KB
    __shared__ __align__(32) bf16 Pst[8][16 * 32];       // 8 KB (per-wave P)

    const int lane = threadIdx.x & 31;
    const int wid  = threadIdx.x >> 5;
    const int bh   = blockIdx.x >> 4;          // 16 blocks of 128 rows per (b,h)
    const int mg   = blockIdx.x & 15;
    const int b    = bh / H;
    const int h    = bh % H;
    const int s0   = mg * 128 + wid * 16;

    const bf16* qrow = Qb + ((size_t)bh * S + s0 + (lane & 15)) * HD + ((lane >> 4) * 16);
    const v16bf qa0 = load_a_frag(qrow);
    const v16bf qa1 = load_a_frag(qrow + 32);

    const bf16* kbase = Kb + (size_t)bh * S * HD;
    const bf16* vbase = Vt + (size_t)bh * HD * S;
    bf16* myP = &Pst[wid][0];

    float mrow[8], lrow[8];
    v8f o[4];
#pragma unroll
    for (int r = 0; r < 8; ++r) { mrow[r] = -3.0e38f; lrow[r] = 0.f; }
#pragma unroll
    for (int j = 0; j < 4; ++j) o[j] = vzero8();

    if (wid == 0) {
        tdm_load_2d(lds_off_of(&Kt[0][0]), kbase, HD, S, HD, 32, HD);
        tdm_load_2d(lds_off_of(&Vs[0][0]), vbase, S, HD, 32, HD, S);
    }

#pragma unroll 1
    for (int t0 = 0; t0 < S; t0 += 32) {
        const int buf = (t0 >> 5) & 1;
        if (wid == 0) __builtin_amdgcn_s_wait_tensorcnt(0);
        __syncthreads();
        if (wid == 0 && (t0 + 32) < S) {
            tdm_load_2d(lds_off_of(&Kt[buf ^ 1][0]),
                        kbase + (size_t)(t0 + 32) * HD, HD, S, HD, 32, HD);
            tdm_load_2d(lds_off_of(&Vs[buf ^ 1][0]),
                        vbase + (t0 + 32), S, HD, 32, HD, S);
        }

        // scores: two 16-col C tiles over 32 keys, K frags from LDS
        const bf16* kt = &Kt[buf][0];
        const int klo = (lane & 15) * 64 + ((lane >> 4) * 16);
        v16bf kb[4];
        kb[0] = *(const v16bf*)(kt + klo);
        kb[1] = *(const v16bf*)(kt + klo + 32);
        kb[2] = *(const v16bf*)(kt + 16 * 64 + klo);
        kb[3] = *(const v16bf*)(kt + 16 * 64 + klo + 32);
        v8f sA = vzero8(), sB = vzero8();
        sA = wmma_bf16(qa0, kb[0], sA);
        sA = wmma_bf16(qa1, kb[1], sA);
        sB = wmma_bf16(qa0, kb[2], sB);
        sB = wmma_bf16(qa1, kb[3], sB);

#pragma unroll
        for (int r = 0; r < 8; ++r) {
            float v = fmaxf(sA[r], sB[r]);
            v = fmaxf(v, __shfl_xor(v, 1, 32));
            v = fmaxf(v, __shfl_xor(v, 2, 32));
            v = fmaxf(v, __shfl_xor(v, 4, 32));
            v = fmaxf(v, __shfl_xor(v, 8, 32));
            const float newm = fmaxf(mrow[r], v);
            const float corr = __expf(mrow[r] - newm);
            const float p0 = __expf(sA[r] - newm);
            const float p1 = __expf(sB[r] - newm);
            float rs = p0 + p1;
            rs += __shfl_xor(rs, 1, 32);
            rs += __shfl_xor(rs, 2, 32);
            rs += __shfl_xor(rs, 4, 32);
            rs += __shfl_xor(rs, 8, 32);
            lrow[r] = lrow[r] * corr + rs;
            mrow[r] = newm;
#pragma unroll
            for (int j = 0; j < 4; ++j) o[j][r] = o[j][r] * corr;
            const int m = r + 8 * (lane >> 4);
            myP[m * 32 + (lane & 15)]      = f2bf(p0);
            myP[m * 32 + 16 + (lane & 15)] = f2bf(p1);
        }

        // per-wave LDS RAW: hardware keeps DS ops in order; block reordering
        asm volatile("" ::: "memory");

        const v16bf pa = load_a_frag(myP + (lane & 15) * 32 + ((lane >> 4) * 16));

        const bf16* vs = &Vs[buf][0];
        const int vo = (lane & 15) * 32 + ((lane >> 4) * 16);
        v16bf vb[4];
#pragma unroll
        for (int j = 0; j < 4; ++j) vb[j] = *(const v16bf*)(vs + j * 16 * 32 + vo);
#pragma unroll
        for (int j = 0; j < 4; ++j) o[j] = wmma_bf16(pa, vb[j], o[j]);
    }

    // normalize + store heads concatenated -> [B,S,D]
#pragma unroll
    for (int r = 0; r < 8; ++r) {
        const float inv = 1.0f / lrow[r];
        const int m = r + 8 * (lane >> 4);
        const size_t orow = ((size_t)(b * S + s0 + m)) * D + h * HD;
#pragma unroll
        for (int j = 0; j < 4; ++j)
            AttO[orow + j * 16 + (lane & 15)] = f2bf(o[j][r] * inv);
    }
}

// ---- kernel 4: output projection, TDM-staged Wo slices ---------------------
// Block = 8 waves sharing one 64-col n-group; wave handles one 16-row m-tile.
__global__ void mha_oproj_kernel(const bf16* __restrict__ AttO,
                                 const bf16* __restrict__ Wob,
                                 const float* __restrict__ bo,
                                 float* __restrict__ out) {
    __shared__ __align__(32) bf16 Wos[2][64 * 32];       // 8 KB

    const int lane = threadIdx.x & 31;
    const int wid  = threadIdx.x >> 5;
    const int ng   = blockIdx.x >> 6;          // 12 n-groups of 64 cols
    const int mt   = (blockIdx.x & 63) * 8 + wid;   // 512 m-tiles
    const int m0   = mt * 16;
    const int n0   = ng * 64;

    const bf16* arow  = AttO + (size_t)(m0 + (lane & 15)) * D + ((lane >> 4) * 16);
    const bf16* wbase = Wob + (size_t)n0 * D;

    v8f c[4];
#pragma unroll
    for (int j = 0; j < 4; ++j) c[j] = vzero8();

    if (wid == 0)
        tdm_load_2d(lds_off_of(&Wos[0][0]), wbase, D, 64, 32, 64, (unsigned long long)D);

#pragma unroll 1
    for (int kk = 0; kk < D; kk += 32) {
        const int buf = (kk >> 5) & 1;
        if (wid == 0) __builtin_amdgcn_s_wait_tensorcnt(0);
        __syncthreads();
        if (wid == 0 && (kk + 32) < D)
            tdm_load_2d(lds_off_of(&Wos[buf ^ 1][0]), wbase + kk + 32,
                        D, 64, 32, 64, (unsigned long long)D);

        v16bf a = load_a_frag(arow + kk);
        const int fo = (lane & 15) * 32 + ((lane >> 4) * 16);
        const bf16* wl = &Wos[buf][0];
        v16bf wb[4];
#pragma unroll
        for (int j = 0; j < 4; ++j) wb[j] = *(const v16bf*)(wl + j * 16 * 32 + fo);
#pragma unroll
        for (int j = 0; j < 4; ++j) c[j] = wmma_bf16(a, wb[j], c[j]);
    }

#pragma unroll
    for (int j = 0; j < 4; ++j) {
        const int col = n0 + j * 16 + (lane & 15);
        const float bv = bo[col];
#pragma unroll
        for (int r = 0; r < 8; ++r) {
            const int m = m0 + r + 8 * (lane >> 4);
            out[(size_t)m * D + col] = c[j][r] + bv;
        }
    }
}

// ---- host launcher ---------------------------------------------------------
extern "C" void kernel_launch(void* const* d_in, const int* in_sizes, int n_in,
                              void* d_out, int out_size, void* d_ws, size_t ws_size,
                              hipStream_t stream) {
    const float* X  = (const float*)d_in[0];
    const float* Wq = (const float*)d_in[1];
    const float* bq = (const float*)d_in[2];
    const float* Wk = (const float*)d_in[3];
    const float* bk = (const float*)d_in[4];
    const float* Wv = (const float*)d_in[5];
    const float* bv = (const float*)d_in[6];
    const float* Wo = (const float*)d_in[7];
    const float* bo = (const float*)d_in[8];
    float* out = (float*)d_out;

    const size_t nX  = (size_t)B * S * D;
    const size_t nW  = (size_t)H * D * HD;
    const size_t nWo = (size_t)D * D;
    const size_t nQ  = (size_t)B * H * S * HD;

    bf16* p    = (bf16*)d_ws;
    bf16* Xbf  = p;              p += nX;
    bf16* Wqt  = p;              p += nW;
    bf16* Wkt  = p;              p += nW;
    bf16* Wvt  = p;              p += nW;
    bf16* Wob  = p;              p += nWo;
    bf16* Qb   = p;              p += nQ;
    bf16* Kb   = p;              p += nQ;
    bf16* Vt   = p;              p += nQ;
    bf16* AttO = p;              p += nX;
    (void)in_sizes; (void)n_in; (void)out_size; (void)ws_size;

    mha_convert_kernel<<<2048, 256, 0, stream>>>(X, Wq, Wk, Wv, Wo,
                                                 Xbf, Wqt, Wkt, Wvt, Wob);
    mha_qkv_kernel<<<768, 256, 0, stream>>>(Xbf, Wqt, Wkt, Wvt,
                                            bq, bk, bv, Qb, Kb, Vt);
    mha_attn_kernel<<<768, 256, 0, stream>>>(Qb, Kb, Vt, AttO);
    mha_oproj_kernel<<<768, 256, 0, stream>>>(AttO, Wob, bo, out);
}
